// L2Similarity_37383395344617
// MI455X (gfx1250) — matrix operand order repaired and checked by reading, not compiled, not run on hardware
//
#include <hip/hip_runtime.h>
#include <hip/hip_bf16.h>
#include <math.h>

// ---------------------------------------------------------------------------
// L2 similarity: out[b,n,m] = -sqrt(max(|x_n|^2 + |y_m|^2 - 2 x_n.y_m, 0))
// B=8, N=M=4096, D=512, fp32 in/out.
// GEMM core: bf16 WMMA with hi/lo fp32 split (3-term) for near-fp32 accuracy.
// ---------------------------------------------------------------------------

typedef __bf16 v16bf __attribute__((ext_vector_type(16)));
typedef __bf16 v8bf  __attribute__((ext_vector_type(8)));
typedef __bf16 v4bf  __attribute__((ext_vector_type(4)));
typedef float  v8f   __attribute__((ext_vector_type(8)));

static constexpr int Dk   = 512;   // K dimension
static constexpr int NN   = 4096;  // lhs rows per batch
static constexpr int MM   = 4096;  // rhs rows per batch
static constexpr int BN   = 128;   // block tile rows (lhs)
static constexpr int BM   = 128;   // block tile cols (rhs)
static constexpr int BK   = 32;    // K step (one bf16 WMMA)
static constexpr int LDSW = 40;    // padded LDS row stride in bf16 (80B -> conflict-free b128)

union Frag { v16bf v; v8bf h[2]; };

__device__ __forceinline__ v8f wmma_bf16(v16bf a, v16bf b, v8f c) {
  // D = A(16x32 bf16) x B(32x16 bf16) + C(16x16 f32)
  return __builtin_amdgcn_wmma_f32_16x16x32_bf16(
      /*neg_a=*/false, a, /*neg_b=*/false, b,
      /*c_mod=*/(short)0, c, /*reuse_a=*/false, /*reuse_b=*/false);
}

// A fragment (16x32, lane = row, K chunks {kk..kk+7} and {kk+16..kk+23}, kk = 8*(lane>>4))
__device__ __forceinline__ v16bf load_frag_a(const __bf16* tile, int lane) {
  const int r  = lane & 15;
  const int kk = (lane >> 4) << 3;     // 0 or 8
  const __bf16* p = tile + r * LDSW + kk;
  Frag f;
  f.h[0] = *(const v8bf*)(p);
  f.h[1] = *(const v8bf*)(p + 16);
  return f.v;
}

// B fragment (32x16, lane = column, K chunks {kk..kk+7} and {kk+16..kk+23}, kk = 8*(lane>>4))
__device__ __forceinline__ v16bf load_frag_b(const __bf16* tile, int lane) {
  const int r  = lane & 15;
  const int kk = (lane >> 4) << 3;     // 0 or 8
  const __bf16* p = tile + r * LDSW + kk;
  Frag f;
  f.h[0] = *(const v8bf*)(p);
  f.h[1] = *(const v8bf*)(p + 16);
  return f.v;
}

// ---------------------------------------------------------------------------
// Pass 1: per-row squared L2 norms (one wave32 per 512-float row)
// ---------------------------------------------------------------------------
__global__ void __launch_bounds__(256)
row_norms_kernel(const float* __restrict__ x, float* __restrict__ o, int nrows) {
  const int lane = threadIdx.x & 31;
  const int wid  = threadIdx.x >> 5;
  const int row  = blockIdx.x * 8 + wid;
  if (row >= nrows) return;
  const float* p = x + (size_t)row * Dk;
  float s = 0.0f;
#pragma unroll
  for (int c = 0; c < 4; ++c) {
    float4 v = *(const float4*)(p + c * 128 + lane * 4);
    s += v.x * v.x + v.y * v.y + v.z * v.z + v.w * v.w;
  }
#pragma unroll
  for (int off = 16; off > 0; off >>= 1) s += __shfl_xor(s, off, 32);
  if (lane == 0) o[row] = s;
}

// ---------------------------------------------------------------------------
// Pass 2: fused GEMM + distance epilogue
// grid = (MM/BM, NN/BN, 8), block = 256 (8 waves in 2x4: wave tile 64x32)
// ---------------------------------------------------------------------------
__global__ void __launch_bounds__(256)
l2sim_gemm_kernel(const float* __restrict__ lhs, const float* __restrict__ rhs,
                  const float* __restrict__ lq,  const float* __restrict__ rq,
                  float* __restrict__ out) {
  __shared__ __bf16 sAh[BN * LDSW];
  __shared__ __bf16 sAl[BN * LDSW];
  __shared__ __bf16 sBh[BM * LDSW];
  __shared__ __bf16 sBl[BM * LDSW];

  const int tid  = threadIdx.x;
  const int lane = tid & 31;
  const int wid  = tid >> 5;
  const int wn   = wid >> 2;      // 0..1  -> 64 lhs rows per wave
  const int wm   = wid & 3;       // 0..3  -> 32 rhs rows per wave

  const int b  = blockIdx.z;
  const int n0 = blockIdx.y * BN;
  const int m0 = blockIdx.x * BM;

  const float* Ab = lhs + (size_t)b * NN * Dk + (size_t)n0 * Dk;
  const float* Bb = rhs + (size_t)b * MM * Dk + (size_t)m0 * Dk;
  const float* lqb = lq + (size_t)b * NN;
  const float* rqb = rq + (size_t)b * MM;
  float* Ob = out + (size_t)b * NN * MM;

  v8f acc[4][2] = {};   // 8 x (16x16) tiles per wave

  // --- register prefetch of first K-slab (128x32 fp32 each, 16 floats/thread) ---
  float4 aR[4], bR[4];
#pragma unroll
  for (int i = 0; i < 4; ++i) {
    const int s   = (i << 8) + tid;     // 0..1023 float4 slots
    const int row = s >> 3;
    const int cg  = (s & 7) << 2;       // element column within the 32-wide slab
    aR[i] = *(const float4*)(Ab + (size_t)row * Dk + cg);
    bR[i] = *(const float4*)(Bb + (size_t)row * Dk + cg);
  }

  for (int ks = 0; ks < Dk / BK; ++ks) {
    __syncthreads();   // all waves finished reading LDS from previous step

    // --- split fp32 -> bf16 hi/lo and stage to LDS ---
#pragma unroll
    for (int i = 0; i < 4; ++i) {
      const int s   = (i << 8) + tid;
      const int row = s >> 3;
      const int cg  = (s & 7) << 2;
      const int off = row * LDSW + cg;
      const float av[4] = {aR[i].x, aR[i].y, aR[i].z, aR[i].w};
      const float bv[4] = {bR[i].x, bR[i].y, bR[i].z, bR[i].w};
      v4bf ah, al, bh, bl;
#pragma unroll
      for (int e = 0; e < 4; ++e) {
        __bf16 h = (__bf16)av[e];
        ah[e] = h;  al[e] = (__bf16)(av[e] - (float)h);
        __bf16 g = (__bf16)bv[e];
        bh[e] = g;  bl[e] = (__bf16)(bv[e] - (float)g);
      }
      *(v4bf*)(sAh + off) = ah;  *(v4bf*)(sAl + off) = al;
      *(v4bf*)(sBh + off) = bh;  *(v4bf*)(sBl + off) = bl;
    }
    __syncthreads();

    // --- prefetch next K-slab while WMMAs run ---
    if (ks + 1 < Dk / BK) {
      const int k0 = (ks + 1) * BK;
#pragma unroll
      for (int i = 0; i < 4; ++i) {
        const int s   = (i << 8) + tid;
        const int row = s >> 3;
        const int cg  = (s & 7) << 2;
        aR[i] = *(const float4*)(Ab + (size_t)row * Dk + k0 + cg);
        bR[i] = *(const float4*)(Bb + (size_t)row * Dk + k0 + cg);
      }
    }

    // --- materialize ALL fragments first so ds_loads drain while WMMAs run ---
    v16bf fb[2][2];   // [j][hi/lo]
    v16bf fa[4][2];   // [i][hi/lo]
#pragma unroll
    for (int j = 0; j < 2; ++j) {
      const int roff = (wm * 32 + j * 16) * LDSW;
      fb[j][0] = load_frag_b(sBh + roff, lane);
      fb[j][1] = load_frag_b(sBl + roff, lane);
    }
#pragma unroll
    for (int i = 0; i < 4; ++i) {
      const int roff = (wn * 64 + i * 16) * LDSW;
      fa[i][0] = load_frag_a(sAh + roff, lane);
      fa[i][1] = load_frag_a(sAl + roff, lane);
    }

    // --- WMMA: acc += Ahi*Bhi + Ahi*Blo + Alo*Bhi ---
#pragma unroll
    for (int i = 0; i < 4; ++i) {
#pragma unroll
      for (int j = 0; j < 2; ++j) {
        acc[i][j] = wmma_bf16(fa[i][0], fb[j][0], acc[i][j]);
        acc[i][j] = wmma_bf16(fa[i][0], fb[j][1], acc[i][j]);
        acc[i][j] = wmma_bf16(fa[i][1], fb[j][0], acc[i][j]);
      }
    }
  }

  // --- epilogue: -sqrt(max(|x|^2 + |y|^2 - 2*cross, 0)) ---
  // C/D layout: VGPR r -> row r (lanes 0-15) / row r+8 (lanes 16-31), col = lane&15
  const int lr = lane & 15;
  const int lh = lane >> 4;
#pragma unroll
  for (int i = 0; i < 4; ++i) {
    const int nbase = n0 + wn * 64 + i * 16 + 8 * lh;
    float lq8[8];
#pragma unroll
    for (int r = 0; r < 8; ++r) lq8[r] = lqb[nbase + r];
#pragma unroll
    for (int j = 0; j < 2; ++j) {
      const int mcol = m0 + wm * 32 + j * 16 + lr;
      const float rqv = rqb[mcol];
#pragma unroll
      for (int r = 0; r < 8; ++r) {
        const float d = lq8[r] + rqv - 2.0f * acc[i][j][r];
        Ob[(size_t)(nbase + r) * MM + mcol] = -sqrtf(fmaxf(d, 0.0f));
      }
    }
  }
}

// ---------------------------------------------------------------------------
extern "C" void kernel_launch(void* const* d_in, const int* in_sizes, int n_in,
                              void* d_out, int out_size, void* d_ws, size_t ws_size,
                              hipStream_t stream) {
  (void)in_sizes; (void)n_in; (void)out_size; (void)ws_size;
  const float* lhs = (const float*)d_in[0];
  const float* rhs = (const float*)d_in[1];
  float* out = (float*)d_out;

  const int Bb = 8;
  float* lq = (float*)d_ws;           // [8*4096]
  float* rq = lq + Bb * NN;           // [8*4096]

  row_norms_kernel<<<(Bb * NN) / 8, 256, 0, stream>>>(lhs, lq, Bb * NN);
  row_norms_kernel<<<(Bb * MM) / 8, 256, 0, stream>>>(rhs, rq, Bb * MM);

  dim3 grid(MM / BM, NN / BN, Bb);
  l2sim_gemm_kernel<<<grid, 256, 0, stream>>>(lhs, rhs, lq, rq, out);
}
